// GCN_1219770712260
// MI455X (gfx1250) — matrix operand order repaired and checked by reading, not compiled, not run on hardware
//
#include <hip/hip_runtime.h>
#include <stdint.h>

// ---------------------------------------------------------------------------
// Types for WMMA fragments / TDM descriptors (CDNA5 / gfx1250, wave32)
// ---------------------------------------------------------------------------
typedef __attribute__((ext_vector_type(16))) __bf16 v16bf;
typedef __attribute__((ext_vector_type(8)))  float  v8f;
typedef __attribute__((ext_vector_type(4)))  unsigned int u32x4;
typedef __attribute__((ext_vector_type(4)))  float  f32x4;
typedef __attribute__((ext_vector_type(4)))  unsigned int v4u;
typedef __attribute__((ext_vector_type(4)))  int    v4i;
typedef __attribute__((ext_vector_type(8)))  int    v8i;

struct Frag256 {            // 32 bytes = 16 bf16 = one WMMA A or B operand
    u32x4 lo;               // VGPRs 0..3
    u32x4 hi;               // VGPRs 4..7
};
static_assert(sizeof(Frag256) == sizeof(v16bf), "frag size mismatch");

__device__ __forceinline__ unsigned short f2bf(float f) {
    // round-to-nearest-even truncation f32 -> bf16
    unsigned int u = __float_as_uint(f);
    unsigned int r = 0x7FFFu + ((u >> 16) & 1u);
    return (unsigned short)((u + r) >> 16);
}

// ---------------------------------------------------------------------------
// Degree / normalization:  dinv[i] = rsqrt(deg_in[i] + 1)
// ---------------------------------------------------------------------------
__global__ __launch_bounds__(256) void deg_count(const int* __restrict__ edst,
                                                 float* __restrict__ deg, int E) {
    int t = blockIdx.x * blockDim.x + threadIdx.x;
    if (t < E) atomicAdd(&deg[edst[t]], 1.0f);
}

__global__ __launch_bounds__(256) void make_dinv(float* __restrict__ deg, int N) {
    int t = blockIdx.x * blockDim.x + threadIdx.x;
    if (t < N) deg[t] = rsqrtf(deg[t] + 1.0f);
}

// ---------------------------------------------------------------------------
// f32 -> bf16 elementwise convert (activations)
// ---------------------------------------------------------------------------
__global__ __launch_bounds__(256) void cvt_bf16(const float* __restrict__ in,
                                                unsigned short* __restrict__ out,
                                                long long n) {
    long long t = (long long)blockIdx.x * blockDim.x + threadIdx.x;
    if (t < n) out[t] = f2bf(in[t]);
}

// Weight convert + transpose: W[k, DOUT] (row-major f32) -> Wt[n, K] (bf16)
// so that a B fragment lane reads a contiguous K-run of one output column.
template <int K, int DOUT>
__global__ __launch_bounds__(256) void cvt_w_t(const float* __restrict__ W,
                                               unsigned short* __restrict__ Wt) {
    int t = blockIdx.x * blockDim.x + threadIdx.x;
    if (t < K * DOUT) {
        int k = t / DOUT, n = t % DOUT;
        Wt[n * K + k] = f2bf(W[t]);
    }
}

// ---------------------------------------------------------------------------
// WMMA GEMM:  H[N, DOUT] = A[N, K] (bf16, row-major) @ Wt[DOUT, K] (bf16)
// Weights staged into LDS once per block via TDM tensor_load_to_lds (builtin
// present on this toolchain; cooperative b128 copy as portable fallback).
// Inner loops are explicitly double-buffered: A fragments one K-step ahead
// (global latency hidden behind 8 WMMAs), B fragments one j-tile ahead
// (ds latency overlapped, waits become dscnt<=2 instead of 0).
// Fragment layouts per CDNA5 ISA 7.12.2 (16-bit A 16x32, B 32x16).
// ---------------------------------------------------------------------------
template <int K, int DOUT>
__global__ __launch_bounds__(256) void gemm_bf16_wmma(
    const unsigned short* __restrict__ A,
    const unsigned short* __restrict__ Wt,
    float* __restrict__ H, int nrows) {
    constexpr int JT  = DOUT / 16;
    constexpr unsigned NEL = (unsigned)DOUT * (unsigned)K;   // bf16 elements
    __shared__ unsigned short sW[DOUT * K];

    // ---- stage Wt[DOUT*K] (contiguous, 2B elements) into LDS ----
#if __has_builtin(__builtin_amdgcn_tensor_load_to_lds)
    if (threadIdx.x == 0) {
        // D# group 0: count=1 | lds_addr=0 | global_addr | type=2
        unsigned long long ga = (unsigned long long)(uintptr_t)Wt;
        v4u g0;
        g0[0] = 1u;                                   // count=1, user mode
        g0[1] = 0u;                                   // lds_addr (sW at offset 0)
        g0[2] = (unsigned)(ga & 0xFFFFFFFFu);         // global_addr[31:0]
        g0[3] = (unsigned)((ga >> 32) & 0x01FFFFFFu)  // global_addr[56:32]
                | 0x80000000u;                        // type=2 ("image")
        // D# group 1: 1-D tile covering the whole weight block
        v8i g1;
        g1[0] = 0x00010000;                           // wg_mask=0, data_size=1 (2B)
        g1[1] = (int)((NEL & 0xFFFFu) << 16);         // tensor_dim0[15:0] @ [31:16]
        g1[2] = (int)((NEL >> 16) & 0xFFFFu)          // tensor_dim0[31:16]
                | (1 << 16);                          // tensor_dim1 = 1
        g1[3] = (int)((NEL & 0xFFFFu) << 16);         // tile_dim0 @ [31:16]
        g1[4] = 0;                                    // tile_dim1=0, tile_dim2=0
        g1[5] = (int)NEL;                             // tensor_dim0_stride[31:0]
        g1[6] = 0;                                    // stride hi, dim1_stride lo
        g1[7] = 0;
        v4i z4 = {0, 0, 0, 0};
#if __clang_major__ >= 23
        v8i z8 = {0, 0, 0, 0, 0, 0, 0, 0};
        __builtin_amdgcn_tensor_load_to_lds(g0, g1, z4, z4, z8, 0);
#else
        __builtin_amdgcn_tensor_load_to_lds(g0, g1, z4, z4, 0);
#endif
        __builtin_amdgcn_s_wait_tensorcnt(0);
    }
#else
    for (unsigned i = threadIdx.x * 8u; i < NEL; i += 256u * 8u)
        *reinterpret_cast<u32x4*>(&sW[i]) =
            *reinterpret_cast<const u32x4*>(Wt + i);
#endif
    __syncthreads();

    const int wave = threadIdx.x >> 5;
    const int lane = threadIdx.x & 31;
    const int tile = blockIdx.x * 8 + wave;     // 16 rows per tile
    if (tile * 16 >= nrows) return;             // wave-uniform: EXEC stays all-1s
    const int row0 = tile * 16;
    const int m  = lane & 15;                   // A row / B col / D col
    const int hi = lane >> 4;

    const unsigned short* arow = A + (size_t)(row0 + m) * K;

    auto loadA = [&](int kk) {
        Frag256 f;
        f.lo = *reinterpret_cast<const u32x4*>(arow + kk + 8 * hi);
        f.hi = *reinterpret_cast<const u32x4*>(arow + kk + 16 + 8 * hi);
        return f;
    };
    auto loadB = [&](int kk, int j) {
        const unsigned short* bcol = sW + (j * 16 + m) * K + kk + 8 * hi;
        Frag256 f;
        f.lo = *reinterpret_cast<const u32x4*>(bcol);
        f.hi = *reinterpret_cast<const u32x4*>(bcol + 16);
        return f;
    };

    v8f acc[JT];
#pragma unroll
    for (int j = 0; j < JT; ++j) acc[j] = (v8f){0, 0, 0, 0, 0, 0, 0, 0};

    Frag256 aCur = loadA(0);
#pragma unroll
    for (int kk = 0; kk < K; kk += 32) {
        Frag256 aNext = aCur;
        if (kk + 32 < K) aNext = loadA(kk + 32);          // global, 1 step ahead
        __builtin_prefetch(arow + kk + 64, 0, 0);         // global_prefetch_b8
        const v16bf av = __builtin_bit_cast(v16bf, aCur);

        Frag256 bCur = loadB(kk, 0);
#pragma unroll
        for (int j = 0; j < JT; ++j) {
            Frag256 bNext = bCur;
            if (j + 1 < JT) bNext = loadB(kk, j + 1);     // LDS, 1 tile ahead
            const v16bf bv = __builtin_bit_cast(v16bf, bCur);
            acc[j] = __builtin_amdgcn_wmma_f32_16x16x32_bf16(
                false, av, false, bv, (short)0, acc[j], false, false);
            bCur = bNext;
        }
        aCur = aNext;
    }

    // D 16x16 f32: lane (n = m), VGPR r -> row (r + 8*hi)
#pragma unroll
    for (int j = 0; j < JT; ++j)
#pragma unroll
        for (int r = 0; r < 8; ++r)
            H[(size_t)(row0 + r + 8 * hi) * DOUT + j * 16 + m] = acc[j][r];
}

// ---------------------------------------------------------------------------
// Edge aggregation: one wave per edge; lanes span DOUT features.
// out[dst, :] += H[src, :] * dinv[src] * dinv[dst]
// ---------------------------------------------------------------------------
template <int DOUT>
__global__ __launch_bounds__(256) void agg_edges(
    const int* __restrict__ esrc, const int* __restrict__ edst,
    const float* __restrict__ dinv, const float* __restrict__ H,
    float* __restrict__ out, int E) {
    const int wave = threadIdx.x >> 5;
    const int lane = threadIdx.x & 31;
    const int e = blockIdx.x * 8 + wave;
    if (e >= E) return;
    const int s = esrc[e];
    const int d = edst[e];
    const float nrm = dinv[s] * dinv[d];
    if constexpr (DOUT == 128) {
        const f32x4 v = *reinterpret_cast<const f32x4*>(H + (size_t)s * 128 + lane * 4);
        float* o = out + (size_t)d * 128 + lane * 4;
        atomicAdd(o + 0, v[0] * nrm);
        atomicAdd(o + 1, v[1] * nrm);
        atomicAdd(o + 2, v[2] * nrm);
        atomicAdd(o + 3, v[3] * nrm);
    } else {  // DOUT == 32
        atomicAdd(out + (size_t)d * DOUT + lane,
                  H[(size_t)s * DOUT + lane] * nrm);
    }
}

// Self-loop + bias:  out[i,c] += H[i,c] * dinv[i]^2 + b[c]
template <int DOUT>
__global__ __launch_bounds__(256) void self_bias(
    const float* __restrict__ H, const float* __restrict__ dinv,
    const float* __restrict__ b, float* __restrict__ out, int N) {
    long long t = (long long)blockIdx.x * blockDim.x + threadIdx.x;
    if (t < (long long)N * DOUT) {
        int i = (int)(t / DOUT);
        int c = (int)(t % DOUT);
        float di = dinv[i];
        out[t] += H[t] * di * di + b[c];
    }
}

// ---------------------------------------------------------------------------
// Row-wise log_softmax over 32 columns: one wave per row (wave32).
// ---------------------------------------------------------------------------
__global__ __launch_bounds__(256) void logsoftmax32(const float* __restrict__ in,
                                                    float* __restrict__ out, int N) {
    const int wave = threadIdx.x >> 5;
    const int lane = threadIdx.x & 31;
    const int row = blockIdx.x * 8 + wave;
    if (row >= N) return;
    float v = in[(size_t)row * 32 + lane];
    float mx = v;
#pragma unroll
    for (int o = 16; o > 0; o >>= 1) mx = fmaxf(mx, __shfl_xor(mx, o, 32));
    float ex = __expf(v - mx);
    float s = ex;
#pragma unroll
    for (int o = 16; o > 0; o >>= 1) s += __shfl_xor(s, o, 32);
    out[(size_t)row * 32 + lane] = v - mx - __logf(s);
}

// ---------------------------------------------------------------------------
// Host orchestration
// ---------------------------------------------------------------------------
static inline int cdiv(long long a, long long b) { return (int)((a + b - 1) / b); }

extern "C" void kernel_launch(void* const* d_in, const int* in_sizes, int n_in,
                              void* d_out, int out_size, void* d_ws, size_t ws_size,
                              hipStream_t stream) {
    const float* x   = (const float*)d_in[0];
    const int*  eix  = (const int*)d_in[1];
    const float* W0  = (const float*)d_in[2];
    const float* b0  = (const float*)d_in[3];
    const float* W1  = (const float*)d_in[4];
    const float* b1  = (const float*)d_in[5];
    const float* W2  = (const float*)d_in[6];
    const float* b2  = (const float*)d_in[7];

    constexpr int IN_DIM = 256, HID = 128, OUT = 32;
    const int N = in_sizes[0] / IN_DIM;
    const int E = in_sizes[1] / 2;
    const int* esrc = eix;          // edge_index row 0
    const int* edst = eix + E;      // edge_index row 1

    // Workspace carve-out (256B aligned slabs)
    char* ws = (char*)d_ws;
    size_t off = 0;
    auto carve = [&](size_t bytes) -> void* {
        off = (off + 255) & ~(size_t)255;
        void* p = ws + off;
        off += bytes;
        return p;
    };
    float*          dinv = (float*)carve((size_t)N * 4);
    unsigned short* Xbf  = (unsigned short*)carve((size_t)N * IN_DIM * 2);
    unsigned short* Wt   = (unsigned short*)carve((size_t)IN_DIM * HID * 2);
    float*          Hbuf = (float*)carve((size_t)N * HID * 4);
    float*          Agg  = (float*)carve((size_t)N * HID * 4);

    const int T = 256;
    const int gemm_grid = cdiv(cdiv(N, 16), 8);
    const int edge_grid = cdiv(E, 8);
    const int row_grid  = cdiv(N, 8);

    // --- normalization ---
    hipMemsetAsync(dinv, 0, (size_t)N * 4, stream);
    deg_count<<<cdiv(E, T), T, 0, stream>>>(edst, dinv, E);
    make_dinv<<<cdiv(N, T), T, 0, stream>>>(dinv, N);

    // --- layer 0: [N,256] @ [256,128] ---
    cvt_bf16<<<cdiv((long long)N * IN_DIM, T), T, 0, stream>>>(x, Xbf, (long long)N * IN_DIM);
    cvt_w_t<IN_DIM, HID><<<cdiv(IN_DIM * HID, T), T, 0, stream>>>(W0, Wt);
    gemm_bf16_wmma<IN_DIM, HID><<<gemm_grid, T, 0, stream>>>(Xbf, Wt, Hbuf, N);
    hipMemsetAsync(Agg, 0, (size_t)N * HID * 4, stream);
    agg_edges<HID><<<edge_grid, T, 0, stream>>>(esrc, edst, dinv, Hbuf, Agg, E);
    self_bias<HID><<<cdiv((long long)N * HID, T), T, 0, stream>>>(Hbuf, dinv, b0, Agg, N);

    // --- layer 1: [N,128] @ [128,128] ---
    cvt_bf16<<<cdiv((long long)N * HID, T), T, 0, stream>>>(Agg, Xbf, (long long)N * HID);
    cvt_w_t<HID, HID><<<cdiv(HID * HID, T), T, 0, stream>>>(W1, Wt);
    gemm_bf16_wmma<HID, HID><<<gemm_grid, T, 0, stream>>>(Xbf, Wt, Hbuf, N);
    hipMemsetAsync(Agg, 0, (size_t)N * HID * 4, stream);
    agg_edges<HID><<<edge_grid, T, 0, stream>>>(esrc, edst, dinv, Hbuf, Agg, E);
    self_bias<HID><<<cdiv((long long)N * HID, T), T, 0, stream>>>(Hbuf, dinv, b1, Agg, N);

    // --- layer 2: [N,128] @ [128,32] ---
    cvt_bf16<<<cdiv((long long)N * HID, T), T, 0, stream>>>(Agg, Xbf, (long long)N * HID);
    cvt_w_t<HID, OUT><<<cdiv(HID * OUT, T), T, 0, stream>>>(W2, Wt);
    gemm_bf16_wmma<HID, OUT><<<gemm_grid, T, 0, stream>>>(Xbf, Wt, Hbuf, N);
    hipMemsetAsync(Agg, 0, (size_t)N * OUT * 4, stream);
    agg_edges<OUT><<<edge_grid, T, 0, stream>>>(esrc, edst, dinv, Hbuf, Agg, E);
    self_bias<OUT><<<cdiv((long long)N * OUT, T), T, 0, stream>>>(Hbuf, dinv, b2, Agg, N);

    // --- log_softmax over 32 cols -> d_out ---
    logsoftmax32<<<row_grid, T, 0, stream>>>(Agg, (float*)d_out, N);
}
